// EdgeGraphConvLayer_58188216926420
// MI455X (gfx1250) — compile-verified, matrix operand order checked by reading, not run on hardware
//
#include <hip/hip_runtime.h>

#define D_NODE 128
#define D_EDGE 64
#define NHEAD  4

typedef __attribute__((ext_vector_type(16))) __bf16 v16bf;
typedef __attribute__((ext_vector_type(8)))  float  v8f;

// Pack a bf16 WMMA A-fragment (16x16x32 layout) from row-major fp32:
// lane holds K-offsets {0..7} at p[0..7] and {16..23} at p[16..23]
// (caller pre-offsets p by +8 for the high lane half).
__device__ __forceinline__ v16bf pack_a_f32(const float* p) {
  const float4 x0 = *(const float4*)(p + 0);
  const float4 x1 = *(const float4*)(p + 4);
  const float4 y0 = *(const float4*)(p + 16);
  const float4 y1 = *(const float4*)(p + 20);
  v16bf a;
  a[0]  = (__bf16)x0.x; a[1]  = (__bf16)x0.y; a[2]  = (__bf16)x0.z; a[3]  = (__bf16)x0.w;
  a[4]  = (__bf16)x1.x; a[5]  = (__bf16)x1.y; a[6]  = (__bf16)x1.z; a[7]  = (__bf16)x1.w;
  a[8]  = (__bf16)y0.x; a[9]  = (__bf16)y0.y; a[10] = (__bf16)y0.z; a[11] = (__bf16)y0.w;
  a[12] = (__bf16)y1.x; a[13] = (__bf16)y1.y; a[14] = (__bf16)y1.z; a[15] = (__bf16)y1.w;
  return a;
}

// IEEE-correct float atomic max via signed-max / unsigned-min trick.
__device__ __forceinline__ void atomic_max_f32(float* addr, float v) {
  if (v >= 0.0f) atomicMax((int*)addr, __float_as_int(v));
  else           atomicMin((unsigned int*)addr, __float_as_uint(v));
}

// ---------------------------------------------------------------- init
__global__ void k_init(float* m, float* dn, float* S, int NH, int NS) {
  const int stride = gridDim.x * blockDim.x;
  for (int i = blockIdx.x * blockDim.x + threadIdx.x; i < NH; i += stride) {
    ((unsigned int*)m)[i] = 0xFF800000u;  // -inf
    dn[i] = 0.0f;
  }
  for (int i = blockIdx.x * blockDim.x + threadIdx.x; i < NS; i += stride)
    S[i] = 0.0f;
}

// ---------------------------------------------------- weight prep (W_w)
// Build bf16 WMMA B-fragments of the stacked weight B[k=h*128+d][dd] =
// W_w[d, h*128+dd], plus bias term 0.25 * sum_h b_w[h*128+dd].
__global__ void k_prep(const float* __restrict__ W_w, const float* __restrict__ b_w,
                       __bf16* __restrict__ Wfrag, float* __restrict__ bterm) {
  const int tid = threadIdx.x;
  for (int idx = tid; idx < 512 * 128; idx += 256) {
    const int k  = idx >> 7, dd = idx & 127;
    const int hh = k >> 7,   d  = k & 127;
    const float v = W_w[d * 512 + hh * 128 + dd];
    const int kb = k >> 5, kr = k & 31;
    const int nt = dd >> 4, nl = dd & 15;
    const int lane = nl + ((kr >> 3) & 1) * 16;
    const int j    = (kr & 7) + ((kr >> 4) << 3);
    Wfrag[(((kb * 8) + nt) * 32 + lane) * 16 + j] = (__bf16)v;
  }
  if (tid < 128) {
    float s = 0.0f;
    for (int hh = 0; hh < NHEAD; ++hh) s += b_w[hh * 128 + tid];
    bterm[tid] = 0.25f * s;
  }
}

// -------------------------------------------------------- edge kernel
// Per wave: 16-edge tile. WMMA edge_feat@W_rel -> LN -> ReLU (wef kept in
// accumulator registers), attention logits u[e,h], atomicMax into m[dst,h].
__global__ __launch_bounds__(256) void k_edge(
    const float* __restrict__ efeat, const float* __restrict__ hfeat,
    const int* __restrict__ src, const int* __restrict__ dst,
    const float* __restrict__ W_rel, const float* __restrict__ b_rel,
    const float* __restrict__ gam, const float* __restrict__ bet,
    const float* __restrict__ W_att, const float* __restrict__ b_att,
    float* __restrict__ u_out, float* __restrict__ m_out, int E) {
  __shared__ v16bf sBv[2 * 8 * 32];                 // W_rel B-fragments (16 KB)
  __shared__ __align__(16) float sWaT[NHEAD * 384]; // W_att^T [h][c]    (6 KB)
  __shared__ float sUwef[8][16][NHEAD];

  __bf16* sB = (__bf16*)sBv;
  const int tid = threadIdx.x;

  // Stage W_rel as bf16 B-fragments (B mirrors A layout with N as lane group).
  for (int idx = tid; idx < D_EDGE * D_NODE; idx += 256) {
    const int k = idx >> 7, n = idx & 127;
    const int kb = k >> 5, kr = k & 31;
    const int nt = n >> 4, nl = n & 15;
    const int lane = nl + ((kr >> 3) & 1) * 16;
    const int j    = (kr & 7) + ((kr >> 4) << 3);
    sB[(((kb * 8) + nt) * 32 + lane) * 16 + j] = (__bf16)W_rel[idx];
  }
  // Stage W_att transposed so per-head columns are contiguous (float4 loads).
  for (int idx = tid; idx < NHEAD * 384; idx += 256) {
    const int hh = idx / 384, c = idx % 384;
    sWaT[idx] = W_att[c * NHEAD + hh];
  }
  __syncthreads();

  const int wave = tid >> 5, lane = tid & 31;
  const int lane15 = lane & 15, hi = lane >> 4;
  const int nTiles = (E + 15) >> 4;
  const int tile = blockIdx.x * 8 + wave;
  if (tile >= nTiles) return;
  const int e0 = tile << 4;

  int eA = e0 + lane15; if (eA >= E) eA = E - 1;
  const v16bf a0 = pack_a_f32(efeat + eA * D_EDGE + 0  + hi * 8);
  const v16bf a1 = pack_a_f32(efeat + eA * D_EDGE + 32 + hi * 8);

  v8f acc[8];
#pragma unroll
  for (int nt = 0; nt < 8; ++nt) {
    v8f c = {};
    v16bf b0 = sBv[(0 * 8 + nt) * 32 + lane];
    v16bf b1 = sBv[(1 * 8 + nt) * 32 + lane];
    c = __builtin_amdgcn_wmma_f32_16x16x32_bf16(false, a0, false, b0, (short)0, c, false, false);
    c = __builtin_amdgcn_wmma_f32_16x16x32_bf16(false, a1, false, b1, (short)0, c, false, false);
    const float br = b_rel[nt * 16 + lane15];
#pragma unroll
    for (int r = 0; r < 8; ++r) acc[nt][r] = c[r] + br;
  }

  // LayerNorm row stats (row m = r + 8*hi, spread over 16 lanes x 8 nt-tiles).
  float s1[8], s2[8];
#pragma unroll
  for (int r = 0; r < 8; ++r) { s1[r] = 0.0f; s2[r] = 0.0f; }
#pragma unroll
  for (int nt = 0; nt < 8; ++nt)
#pragma unroll
    for (int r = 0; r < 8; ++r) { const float x = acc[nt][r]; s1[r] += x; s2[r] += x * x; }
#pragma unroll
  for (int mask = 1; mask <= 8; mask <<= 1)
#pragma unroll
    for (int r = 0; r < 8; ++r) {
      s1[r] += __shfl_xor(s1[r], mask, 32);
      s2[r] += __shfl_xor(s2[r], mask, 32);
    }
  float mu[8], rs[8];
#pragma unroll
  for (int r = 0; r < 8; ++r) {
    mu[r] = s1[r] * (1.0f / 128.0f);
    const float var = s2[r] * (1.0f / 128.0f) - mu[r] * mu[r];
    rs[r] = rsqrtf(var + 1e-5f);
  }
#pragma unroll
  for (int nt = 0; nt < 8; ++nt) {
    const float g = gam[nt * 16 + lane15], b = bet[nt * 16 + lane15];
#pragma unroll
    for (int r = 0; r < 8; ++r) {
      const float x = (acc[nt][r] - mu[r]) * rs[r] * g + b;
      acc[nt][r] = x > 0.0f ? x : 0.0f;  // wef (stays in registers)
    }
  }

  // Attention contribution from wef: dot(wef_row, W_att[256+.., h]).
  float pw[8][NHEAD];
#pragma unroll
  for (int r = 0; r < 8; ++r)
#pragma unroll
    for (int hh = 0; hh < NHEAD; ++hh) pw[r][hh] = 0.0f;
#pragma unroll
  for (int nt = 0; nt < 8; ++nt)
#pragma unroll
    for (int hh = 0; hh < NHEAD; ++hh) {
      const float wv = sWaT[hh * 384 + 256 + nt * 16 + lane15];
#pragma unroll
      for (int r = 0; r < 8; ++r) pw[r][hh] += acc[nt][r] * wv;
    }
#pragma unroll
  for (int mask = 1; mask <= 8; mask <<= 1)
#pragma unroll
    for (int r = 0; r < 8; ++r)
#pragma unroll
      for (int hh = 0; hh < NHEAD; ++hh)
        pw[r][hh] += __shfl_xor(pw[r][hh], mask, 32);
  if (lane15 == 0) {
#pragma unroll
    for (int r = 0; r < 8; ++r)
#pragma unroll
      for (int hh = 0; hh < NHEAD; ++hh)
        sUwef[wave][r + 8 * hi] [hh] = pw[r][hh];
  }

  // Gathered src/dst dot products: 2 lanes per edge, 64 cols each.
  const int eg = e0 + lane15;
  const bool valid = (eg < E);
  const int ec = valid ? eg : E - 1;
  const int si = src[ec], di = dst[ec];
  const float* hs = hfeat + si * D_NODE + hi * 64;
  const float* hd = hfeat + di * D_NODE + hi * 64;
  float p[NHEAD] = {0.0f, 0.0f, 0.0f, 0.0f};
  for (int j = 0; j < 64; j += 4) {
    const float4 d4 = *(const float4*)(hd + j);
    const float4 s4 = *(const float4*)(hs + j);
    const int c = hi * 64 + j;
#pragma unroll
    for (int hh = 0; hh < NHEAD; ++hh) {
      const float4 wd = *(const float4*)&sWaT[hh * 384 + c];
      const float4 ws = *(const float4*)&sWaT[hh * 384 + 128 + c];
      p[hh] += d4.x * wd.x + d4.y * wd.y + d4.z * wd.z + d4.w * wd.w
             + s4.x * ws.x + s4.y * ws.y + s4.z * ws.z + s4.w * ws.w;
    }
  }
#pragma unroll
  for (int hh = 0; hh < NHEAD; ++hh) p[hh] += __shfl_xor(p[hh], 16, 32);

  if (hi == 0 && valid) {
#pragma unroll
    for (int hh = 0; hh < NHEAD; ++hh) {
      const float t = p[hh] + sUwef[wave][lane15][hh] + b_att[hh];
      const float u = t > 0.0f ? t : 0.2f * t;  // leaky_relu
      u_out[eg * NHEAD + hh] = u;
      atomic_max_f32(&m_out[di * NHEAD + hh], u);
    }
  }
}

// -------------------------------------------------- softmax denominator
__global__ void k_denom(float* __restrict__ u_buf, const float* __restrict__ m_buf,
                        const int* __restrict__ dst, float* __restrict__ dn_buf, int E4) {
  const int i = blockIdx.x * blockDim.x + threadIdx.x;
  if (i >= E4) return;
  const int e = i >> 2, hh = i & 3;
  const int n = dst[e];
  const float t = __expf(u_buf[i] - m_buf[n * 4 + hh]);
  u_buf[i] = t;  // rewritten by k_edge every launch -> deterministic
  atomicAdd(&dn_buf[n * 4 + hh], t);
}

// ------------------------------------------- alpha-weighted scatter-sum
// One wave per edge: S[dst, h*128+d] += alpha[e,h] * h_src[e,d].
__global__ __launch_bounds__(256) void k_scatter(
    const float* __restrict__ t_buf, const float* __restrict__ dn_buf,
    const int* __restrict__ src, const int* __restrict__ dst,
    const float* __restrict__ hfeat, float* __restrict__ S, int E) {
  const int wave = threadIdx.x >> 5, lane = threadIdx.x & 31;
  const int e = blockIdx.x * 8 + wave;
  if (e >= E) return;
  const int si = src[e], di = dst[e];
  const float4 hs = *(const float4*)(hfeat + si * 128 + lane * 4);
  const float4 t4 = *(const float4*)(t_buf + e * 4);
  const float4 dn = *(const float4*)(dn_buf + di * 4);
  const float al[4] = {t4.x / dn.x, t4.y / dn.y, t4.z / dn.z, t4.w / dn.w};
  float* base = S + di * 512 + lane * 4;
#pragma unroll
  for (int hh = 0; hh < 4; ++hh) {
    const float a = al[hh];
    float* bp = base + hh * 128;
    atomicAdd(bp + 0, a * hs.x);
    atomicAdd(bp + 1, a * hs.y);
    atomicAdd(bp + 2, a * hs.z);
    atomicAdd(bp + 3, a * hs.w);
  }
}

// ------------------------------------------- node GEMM + PReLU + mask
// out = prelu( (S @ Wstack) / H + bterm ), masked to h where deg==0.
__global__ __launch_bounds__(256) void k_nodegemm(
    const float* __restrict__ S, const __bf16* __restrict__ Wfrag,
    const float* __restrict__ bterm, const float* __restrict__ dn_buf,
    const float* __restrict__ h_in, const float* __restrict__ prelu_a,
    float* __restrict__ out, int N) {
  const int wave = threadIdx.x >> 5, lane = threadIdx.x & 31;
  const int lane15 = lane & 15, hi = lane >> 4;
  const int n0 = blockIdx.x * 16;
  const int nt = wave;  // column tile (8 waves x 16 cols = 128)
  int rowA = n0 + lane15; if (rowA >= N) rowA = N - 1;
  v8f acc = {};
#pragma unroll 4
  for (int kb = 0; kb < 16; ++kb) {
    const v16bf a = pack_a_f32(S + rowA * 512 + kb * 32 + hi * 8);
    const v16bf b = *(const v16bf*)(Wfrag + ((kb * 8 + nt) * 32 + lane) * 16);
    acc = __builtin_amdgcn_wmma_f32_16x16x32_bf16(false, a, false, b, (short)0, acc, false, false);
  }
  const int col = nt * 16 + lane15;
  const float bt = bterm[col];
  const float ap = prelu_a[0];
#pragma unroll
  for (int r = 0; r < 8; ++r) {
    const int n = n0 + r + 8 * hi;
    if (n < N) {
      float v = acc[r] * 0.25f + bt;
      v = v >= 0.0f ? v : ap * v;
      out[n * 128 + col] = (dn_buf[n * 4] > 0.0f) ? v : h_in[n * 128 + col];
    }
  }
}

// -------------------------------------------------------------- launch
static inline size_t align_up256(size_t x) { return (x + 255) & ~(size_t)255; }

extern "C" void kernel_launch(void* const* d_in, const int* in_sizes, int n_in,
                              void* d_out, int out_size, void* d_ws, size_t ws_size,
                              hipStream_t stream) {
  const float* hfeat   = (const float*)d_in[0];
  const float* efeat   = (const float*)d_in[1];
  const int*   src     = (const int*)  d_in[2];
  const int*   dst     = (const int*)  d_in[3];
  const float* W_rel   = (const float*)d_in[4];
  const float* b_rel   = (const float*)d_in[5];
  const float* gam     = (const float*)d_in[6];
  const float* bet     = (const float*)d_in[7];
  const float* W_att   = (const float*)d_in[8];
  const float* b_att   = (const float*)d_in[9];
  const float* W_w     = (const float*)d_in[10];
  const float* b_w     = (const float*)d_in[11];
  const float* prelu_a = (const float*)d_in[12];
  const int N = in_sizes[0] / D_NODE;
  const int E = in_sizes[2];

  char* w = (char*)d_ws;
  float*  u_buf  = (float*)w;  w += align_up256((size_t)E * NHEAD * sizeof(float));
  float*  m_buf  = (float*)w;  w += align_up256((size_t)N * NHEAD * sizeof(float));
  float*  dn_buf = (float*)w;  w += align_up256((size_t)N * NHEAD * sizeof(float));
  float*  S_buf  = (float*)w;  w += align_up256((size_t)N * 512 * sizeof(float));
  __bf16* Wfrag  = (__bf16*)w; w += align_up256((size_t)512 * 128 * sizeof(__bf16));
  float*  bterm  = (float*)w;

  k_init<<<2048, 256, 0, stream>>>(m_buf, dn_buf, S_buf, N * NHEAD, N * 512);
  k_prep<<<1, 256, 0, stream>>>(W_w, b_w, Wfrag, bterm);
  const int nTiles = (E + 15) / 16;
  k_edge<<<(nTiles + 7) / 8, 256, 0, stream>>>(efeat, hfeat, src, dst, W_rel, b_rel,
                                               gam, bet, W_att, b_att, u_buf, m_buf, E);
  k_denom<<<(E * NHEAD + 255) / 256, 256, 0, stream>>>(u_buf, m_buf, dst, dn_buf, E * NHEAD);
  k_scatter<<<(E + 7) / 8, 256, 0, stream>>>(u_buf, dn_buf, src, dst, hfeat, S_buf, E);
  k_nodegemm<<<(N + 15) / 16, 256, 0, stream>>>(S_buf, Wfrag, bterm, dn_buf, hfeat,
                                                prelu_a, (float*)d_out, N);
  (void)n_in; (void)out_size; (void)ws_size;
}
